// model_33792802685177
// MI455X (gfx1250) — matrix operand order repaired and checked by reading, not compiled, not run on hardware
//
#include <hip/hip_runtime.h>
#include <hip/hip_bf16.h>

// ---------------------------------------------------------------------------
// DeepAR-style 4-layer LSTM, B=512, H=1024, T=256 (+64 autoregressive steps).
// bf16 WMMA GEMM (f32 accum), gate-interleaved L2-resident packed weights,
// fused LSTM-cell epilogue.  h state is kept in bf16 (the precision the GEMM
// consumes anyway), so BOTH GEMM operands stream straight into LDS via
// double-buffered GLOBAL_LOAD_ASYNC_TO_LDS_B128 (ASYNCcnt): the inner loop is
// s_wait_asynccnt -> barrier -> ds_load_b128 fragments -> 8x WMMA -> barrier
// -> async refill, with no VGPR staging and no global-latency exposure.
// ---------------------------------------------------------------------------

#define B_   512
#define T_   256
#define TT_  320          // T + future
#define H_   1024
#define NG_  4096         // 4*H gates
#define COV_ 7

typedef __attribute__((ext_vector_type(16))) __bf16 v16bf;
typedef __attribute__((ext_vector_type(8)))  float  v8f;

union AFrag { v16bf v; unsigned u[8]; };

static __device__ __forceinline__ unsigned short f2bf(float f) {
    union { float f; unsigned u; } x; x.f = f;
    unsigned r = x.u + 0x7FFFu + ((x.u >> 16) & 1u);   // round-to-nearest-even
    return (unsigned short)(r >> 16);
}
static __device__ __forceinline__ float bf2f(unsigned short u) {
    union { unsigned u; float f; } x; x.u = ((unsigned)u) << 16; return x.f;
}
static __device__ __forceinline__ float sigmoidf_(float x) {
    return 1.0f / (1.0f + __expf(-x));
}

// Raw LDS byte offset of a __shared__ object (generic -> addrspace(3) ptrtoint)
static __device__ __forceinline__ unsigned lds_off(const void* p) {
    return (unsigned)(unsigned long long)
        (__attribute__((address_space(3))) const void*)p;
}

#if __has_builtin(__builtin_amdgcn_s_wait_asynccnt)
#define WAIT_ASYNC(n) __builtin_amdgcn_s_wait_asynccnt(n)
#else
#define WAIT_ASYNC(n) asm volatile("s_wait_asynccnt %0" :: "i"(n) : "memory")
#endif

// ---------------------------------------------------------------------------
// Weight packing: gate-interleave rows so packed column n' = 4*j + g maps to
// original gate row r = g*H + j (gate order i,f,g,o).  Layout [NG][K] bf16.
// ---------------------------------------------------------------------------
__global__ __launch_bounds__(256) void pack_layer0(
    const float* __restrict__ W_ih0, const float* __restrict__ W_hh0,
    const float* __restrict__ b_ih0, const float* __restrict__ b_hh0,
    unsigned short* __restrict__ whh0, float* __restrict__ wih0,
    float* __restrict__ bias0)
{
    const int n = blockIdx.x;
    const int r = (n & 3) * H_ + (n >> 2);
    for (int k = threadIdx.x; k < H_; k += 256)
        whh0[(size_t)n * H_ + k] = f2bf(W_hh0[(size_t)r * H_ + k]);
    if (threadIdx.x < 8)
        wih0[(size_t)n * 8 + threadIdx.x] = W_ih0[(size_t)r * 8 + threadIdx.x];
    if (threadIdx.x == 0)
        bias0[n] = b_ih0[r] + b_hh0[r];
}

__global__ __launch_bounds__(256) void pack_rec(
    const float* __restrict__ W_ih_r, const float* __restrict__ W_hh_r,
    const float* __restrict__ b_ih_r, const float* __restrict__ b_hh_r,
    unsigned short* __restrict__ wcat, float* __restrict__ biasr)
{
    const int n = blockIdx.x;
    const int l = blockIdx.y;
    const int r = (n & 3) * H_ + (n >> 2);
    const float* ih = W_ih_r + ((size_t)l * NG_ + r) * H_;
    const float* hh = W_hh_r + ((size_t)l * NG_ + r) * H_;
    unsigned short* row = wcat + ((size_t)l * NG_ + n) * (2 * H_);
    for (int k = threadIdx.x; k < H_; k += 256) {
        row[k]      = f2bf(ih[k]);
        row[H_ + k] = f2bf(hh[k]);
    }
    if (threadIdx.x == 0)
        biasr[(size_t)l * NG_ + n] = b_ih_r[(size_t)l * NG_ + r] + b_hh_r[(size_t)l * NG_ + r];
}

__global__ __launch_bounds__(256) void zero_f32(float* __restrict__ p, size_t n) {
    size_t i = (size_t)blockIdx.x * blockDim.x + threadIdx.x;
    size_t stride = (size_t)gridDim.x * blockDim.x;
    for (; i < n; i += stride) p[i] = 0.0f;
}

// ---------------------------------------------------------------------------
// Fused LSTM layer kernel.  Grid (NG/128, B/128), block 256 = 8 wave32s
// (2M x 4N).  Each wave: 64x32 tile = 4x2 16x16 WMMA accumulators.
// Both A (bf16 h-state) and B (bf16 weights) tiles streamed into LDS with
// double-buffered async global->LDS copies; 80B row stride keeps every
// 16B async store and every ds_load_b128 fragment read aligned and
// bank-conflict-free.
// ---------------------------------------------------------------------------
__global__ __launch_bounds__(256) void lstm_layer(
    const unsigned short* __restrict__ hprev,   // [B][H] bf16
    const unsigned short* __restrict__ hbelow,  // [B][H] bf16
    const unsigned short* __restrict__ Wpk,     // [NG][K] bf16
    const float* __restrict__ bias,             // [NG]
    const float* __restrict__ Wih0,             // [NG][8] (layer0 only)
    const float* __restrict__ seq_in,           // [B][T]
    const float* __restrict__ cov,              // [B][TT][7]
    const float* __restrict__ outs,             // [B][TT] (d_out)
    unsigned short* __restrict__ hout,          // [B][H] bf16
    float* __restrict__ cstate,                 // [B][H] f32
    int K, int t, int is_layer0)
{
    __shared__ unsigned As[2][128 * 20];   // A dbl-buf: 128 M-rows x 64B (+pad)
    __shared__ unsigned Bs[2][128 * 20];   // B dbl-buf: 128 N-rows x 64B (+pad)
    __shared__ float    Gs[64 * 129];      // epilogue gate staging (half tile)

    const int tid   = threadIdx.x;
    const int wave  = tid >> 5;
    const int lane  = tid & 31;
    const int lh    = lane >> 4;
    const int l15   = lane & 15;
    const int waveM = wave >> 2;           // 0..1
    const int waveN = wave & 3;            // 0..3
    const int mBlk  = blockIdx.y * 128;
    const int nBlk  = blockIdx.x * 128;
    const int nch   = K >> 5;              // number of 32-wide K chunks

    const unsigned asBase[2] = { lds_off(&As[0][0]), lds_off(&As[1][0]) };
    const unsigned bsBase[2] = { lds_off(&Bs[0][0]), lds_off(&Bs[1][0]) };

    // Async-copy this wave's 16-row slice of a 128x64B tile into LDS.
    // 2 instructions x 32 lanes x 16B = 1KB per wave; 8 waves cover the tile.
    auto issue_tile = [&](const void* gbase, size_t rowStrideB, unsigned ldsBase) {
        #pragma unroll
        for (int j = 0; j < 2; ++j) {
            const int idx = j * 32 + lane;
            const int r   = (wave << 4) + (idx >> 2);
            const int c16 = idx & 3;
            unsigned long long g = (unsigned long long)gbase
                                 + (size_t)r * rowStrideB + (unsigned)(c16 * 16);
            unsigned l = ldsBase + (unsigned)(r * 80 + c16 * 16);
            asm volatile("global_load_async_to_lds_b128 %0, %1, off"
                         :: "v"(l), "v"(g) : "memory");
        }
    };
    // Issue both operands of chunk `ic` into buffer `buf`.
    auto issue_chunk = [&](int ic, int buf) {
        const int kc = ic << 5;
        const unsigned short* asrc = (kc < H_)
            ? (hbelow + (size_t)mBlk * H_ + kc)
            : (hprev  + (size_t)mBlk * H_ + (kc - H_));
        issue_tile(asrc, (size_t)H_ * 2, asBase[buf]);
        issue_tile(Wpk + (size_t)nBlk * K + kc, (size_t)K * 2, bsBase[buf]);
    };

    v8f acc[4][2];
    #pragma unroll
    for (int i = 0; i < 4; ++i)
        #pragma unroll
        for (int j = 0; j < 2; ++j)
            acc[i][j] = (v8f){0.f,0.f,0.f,0.f,0.f,0.f,0.f,0.f};

    // Prologue: two chunks in flight (8 async instructions outstanding).
    issue_chunk(0, 0);
    if (nch > 1) issue_chunk(1, 1);

    for (int ic = 0; ic < nch; ++ic) {
        const int buf = ic & 1;

        // Wait for this wave's slice of chunk ic (ASYNCcnt is in-order).
        if (ic + 1 < nch) WAIT_ASYNC(4); else WAIT_ASYNC(0);
        __syncthreads();   // all waves' slices of chunk ic are in LDS

        // ---- WMMA on chunk ic: contiguous b128 fragment reads ----
        AFrag bfr[2];
        #pragma unroll
        for (int tj = 0; tj < 2; ++tj) {
            const int n = waveN * 32 + tj * 16 + l15;
            const unsigned* bp = &Bs[buf][n * 20 + lh * 8];
            #pragma unroll
            for (int v = 0; v < 8; ++v) bfr[tj].u[v] = bp[v];
        }
        #pragma unroll
        for (int ti = 0; ti < 4; ++ti) {
            const int row = waveM * 64 + ti * 16 + l15;
            const unsigned* ap = &As[buf][row * 20 + lh * 8];
            AFrag af;
            #pragma unroll
            for (int v = 0; v < 8; ++v)
                af.u[v] = ap[v];
            acc[ti][0] = __builtin_amdgcn_wmma_f32_16x16x32_bf16(
                false, af.v, false, bfr[0].v, (short)0, acc[ti][0], false, false);
            acc[ti][1] = __builtin_amdgcn_wmma_f32_16x16x32_bf16(
                false, af.v, false, bfr[1].v, (short)0, acc[ti][1], false, false);
        }
        __syncthreads();   // everyone done reading As[buf]/Bs[buf]

        // Refill this buffer for chunk ic+2 (consumed 2 iterations later).
        if (ic + 2 < nch) issue_chunk(ic + 2, buf);
    }

    // ---- fused LSTM cell epilogue (two 64-row passes through LDS) ----
    for (int pass = 0; pass < 2; ++pass) {
        if (waveM == pass) {
            #pragma unroll
            for (int ti = 0; ti < 4; ++ti)
                #pragma unroll
                for (int tj = 0; tj < 2; ++tj) {
                    const int ml = ti * 16 + lh * 8;
                    const int nc = waveN * 32 + tj * 16 + l15;
                    #pragma unroll
                    for (int r = 0; r < 8; ++r)
                        Gs[(ml + r) * 129 + nc] = acc[ti][tj][r];
                }
        }
        __syncthreads();

        for (int idx = tid; idx < 64 * 32; idx += 256) {
            const int ml = idx >> 5;
            const int j  = idx & 31;
            const int mg = mBlk + pass * 64 + ml;
            const int ug = (nBlk >> 2) + j;
            float g4[4];
            #pragma unroll
            for (int g = 0; g < 4; ++g)
                g4[g] = Gs[ml * 129 + 4 * j + g] + bias[nBlk + 4 * j + g];

            if (is_layer0) {
                float x[8];
                x[0] = (t < T_) ? seq_in[(size_t)mg * T_ + t]
                                : outs[(size_t)mg * TT_ + (t - 1)];
                #pragma unroll
                for (int k = 1; k < 8; ++k)
                    x[k] = cov[((size_t)mg * TT_ + t) * COV_ + (k - 1)];
                #pragma unroll
                for (int g = 0; g < 4; ++g) {
                    const float* w = Wih0 + (size_t)(nBlk + 4 * j + g) * 8;
                    float s = 0.f;
                    #pragma unroll
                    for (int k = 0; k < 8; ++k) s += x[k] * w[k];
                    g4[g] += s;
                }
            }
            const float si = sigmoidf_(g4[0]);
            const float sf = sigmoidf_(g4[1]);
            const float tg = tanhf(g4[2]);
            const float so = sigmoidf_(g4[3]);
            const float co = cstate[(size_t)mg * H_ + ug];
            const float cn = sf * co + si * tg;
            const float hn = so * tanhf(cn);
            cstate[(size_t)mg * H_ + ug] = cn;
            hout  [(size_t)mg * H_ + ug] = f2bf(hn);
        }
        __syncthreads();
    }
}

// ---------------------------------------------------------------------------
// Output projection: out[b,t] = h3[b,:] . Wl + bl  (one wave32 per batch row)
// ---------------------------------------------------------------------------
__global__ __launch_bounds__(256) void proj_kernel(
    const unsigned short* __restrict__ h3, const float* __restrict__ Wl,
    const float* __restrict__ bl, float* __restrict__ outs, int t)
{
    const int wave = threadIdx.x >> 5;
    const int lane = threadIdx.x & 31;
    const int b = blockIdx.x * 8 + wave;
    const unsigned short* hb = h3 + (size_t)b * H_;
    float s = 0.f;
    for (int k = lane; k < H_; k += 32) s += bf2f(hb[k]) * Wl[k];
    #pragma unroll
    for (int off = 16; off; off >>= 1) s += __shfl_down(s, off, 32);
    if (lane == 0) outs[(size_t)b * TT_ + t] = s + bl[0];
}

// ---------------------------------------------------------------------------
extern "C" void kernel_launch(void* const* d_in, const int* in_sizes, int n_in,
                              void* d_out, int out_size, void* d_ws, size_t ws_size,
                              hipStream_t stream) {
    (void)in_sizes; (void)n_in; (void)out_size; (void)ws_size;

    const float* input  = (const float*)d_in[0];
    const float* cov    = (const float*)d_in[1];
    const float* W_ih0  = (const float*)d_in[2];
    const float* W_hh0  = (const float*)d_in[3];
    const float* b_ih0  = (const float*)d_in[4];
    const float* b_hh0  = (const float*)d_in[5];
    const float* W_ih_r = (const float*)d_in[6];
    const float* W_hh_r = (const float*)d_in[7];
    const float* b_ih_r = (const float*)d_in[8];
    const float* b_hh_r = (const float*)d_in[9];
    const float* Wl     = (const float*)d_in[10];
    const float* bl     = (const float*)d_in[11];
    float* outs = (float*)d_out;

    char* ws = (char*)d_ws;
    const size_t S_WHH0 = (size_t)NG_ * H_ * 2;          // 8 MB bf16
    const size_t S_WCAT = (size_t)NG_ * 2 * H_ * 2;      // 16 MB bf16 / layer
    const size_t S_WIH0 = (size_t)NG_ * 8 * 4;
    const size_t S_B0   = (size_t)NG_ * 4;
    const size_t S_BR   = (size_t)3 * NG_ * 4;
    const size_t S_HBF  = (size_t)4 * B_ * H_ * 2;       // 4 MB bf16 h set
    const size_t S_C    = (size_t)4 * B_ * H_ * 4;       // 8 MB f32 c
    size_t off = 0;
    unsigned short* whh0 = (unsigned short*)(ws + off); off += S_WHH0;
    unsigned short* wcat = (unsigned short*)(ws + off); off += 3 * S_WCAT;
    float* wih0  = (float*)(ws + off); off += S_WIH0;
    float* bias0 = (float*)(ws + off); off += S_B0;
    float* biasr = (float*)(ws + off); off += S_BR;
    unsigned short* hA = (unsigned short*)(ws + off); off += S_HBF;
    unsigned short* hB = (unsigned short*)(ws + off); off += S_HBF;
    float* cS = (float*)(ws + off); off += S_C;

    pack_layer0<<<NG_, 256, 0, stream>>>(W_ih0, W_hh0, b_ih0, b_hh0, whh0, wih0, bias0);
    pack_rec<<<dim3(NG_, 3), 256, 0, stream>>>(W_ih_r, W_hh_r, b_ih_r, b_hh_r, wcat, biasr);
    // hA, hB, cS are contiguous: zero all state (16 MB) in one pass.
    zero_f32<<<2048, 256, 0, stream>>>((float*)hA, (2 * S_HBF + S_C) / 4);

    const dim3 grid(NG_ / 128, B_ / 128);
    const size_t LSTRIDE = (size_t)B_ * H_;
    for (int t = 0; t < TT_; ++t) {
        unsigned short* hin  = (t & 1) ? hB : hA;
        unsigned short* hnew = (t & 1) ? hA : hB;
        lstm_layer<<<grid, 256, 0, stream>>>(
            hin, hin, whh0, bias0, wih0,
            input, cov, outs, hnew, cS, H_, t, 1);
        for (int l = 1; l < 4; ++l) {
            lstm_layer<<<grid, 256, 0, stream>>>(
                hin + (size_t)l * LSTRIDE,
                hnew + (size_t)(l - 1) * LSTRIDE,
                wcat + (size_t)(l - 1) * NG_ * 2 * H_,
                biasr + (size_t)(l - 1) * NG_, wih0,
                input, cov, outs,
                hnew + (size_t)l * LSTRIDE,
                cS + (size_t)l * LSTRIDE,
                2 * H_, t, 0);
        }
        proj_kernel<<<B_ / 8, 256, 0, stream>>>(
            hnew + (size_t)3 * LSTRIDE, Wl, bl, outs, t);
    }
}
